// GAT_77154792506039
// MI455X (gfx1250) — matrix operand order repaired and checked by reading, compiled, not run on hardware
//
#include <hip/hip_runtime.h>
#include <stdint.h>

// ---------------------------------------------------------------------------
// GAT layer, MI455X (gfx1250, wave32, WMMA bf16 16x16x32).
// batch_e is dead code in the reference (mask is 0.9/0.1, both > 0), so the
// softmax is unmasked and rank-1 structured: e[i,j] = lrelu(s1_i + s2_j).
// Memory-bound (~130 MB essential HBM traffic vs 30 GFLOP), so bf16 WMMA
// with f32 accumulation + f32 softmax stats is the right precision split.
// ---------------------------------------------------------------------------

#define B_   32
#define N_   1024
#define E_   768
#define OUT_ 256
#define ALPHA 0.2f

typedef __bf16 bf16;
typedef __attribute__((ext_vector_type(16))) __bf16 v16bf;
typedef __attribute__((ext_vector_type(8)))  float  v8f;

union Frag  { v16bf v; uint4 q[2]; };
union Pack16 { bf16 h[16]; uint4 q[2]; };

__device__ __forceinline__ float lrelu(float x) { return fmaxf(x, ALPHA * x); }

// Async global->LDS copy, 16B per lane (ASYNCcnt path). Low 32 bits of a
// generic pointer to __shared__ == LDS byte offset (AS3 offset).
__device__ __forceinline__ void async_copy_b128(const void* gptr, void* lptr) {
    unsigned lds = (unsigned)(uintptr_t)lptr;
    asm volatile("global_load_async_to_lds_b128 %0, %1, off"
                 :: "v"(lds), "v"(gptr) : "memory");
}
__device__ __forceinline__ void wait_async0() {
    asm volatile("s_wait_asynccnt 0x0" ::: "memory");
}

#define LDH 40   // LDS row stride in halves (80B: 16B-aligned, bank-skewed)

// ---------------------------------------------------------------------------
// Kernel 0: convert W_w -> bf16 (hardware cvt), zero pooled accumulator.
// ---------------------------------------------------------------------------
__global__ __launch_bounds__(256)
void gat_prep(const float* __restrict__ Ww, bf16* __restrict__ Wbf,
              float* __restrict__ pooled) {
    int i = blockIdx.x * 256 + threadIdx.x;
    if (i < OUT_ * E_) Wbf[i] = (bf16)Ww[i];
    if (i < B_ * OUT_) pooled[i] = 0.0f;
}

// ---------------------------------------------------------------------------
// Kernel 1: Wh[r,o] = sum_e sent[r,e]*W[o,e] + b[o]   (r = b*N+n, 32768 rows)
// WG tile 128(M) x 64(O), 8 waves (4x2), each wave 32x32 = 4 WMMA tiles.
// A tile: f32->bf16 cvt fused into LDS staging (+ prefetch of next k-slab).
// W tile: pure copy -> async global->LDS (ASYNCcnt).
// ---------------------------------------------------------------------------
__global__ __launch_bounds__(256)
void gat_gemm1(const float* __restrict__ A, const bf16* __restrict__ Wbf,
               const float* __restrict__ Wb, bf16* __restrict__ Whbf) {
    __shared__ bf16 lA[128 * LDH];
    __shared__ bf16 lW[64 * LDH];
    const int tid    = threadIdx.x;
    const int lane   = tid & 31;
    const int wave   = tid >> 5;
    const int mW     = wave & 3;        // 0..3  -> 32-row strip
    const int oW     = wave >> 2;       // 0..1  -> 32-col strip
    const int lan    = lane & 15;
    const int laneHi = lane >> 4;
    const int mBlk   = blockIdx.x * 128;
    const int oBlk   = blockIdx.y * 64;

    v8f acc[2][2] = {};

    for (int k0 = 0; k0 < E_; k0 += 32) {
        // ---- W tile: 64 o x 32 k bf16, straight async copy (16B/lane)
        {
            int o  = tid >> 2;
            int kp = (tid & 3) * 8;
            async_copy_b128(Wbf + (size_t)(oBlk + o) * E_ + k0 + kp,
                            &lW[o * LDH + kp]);
        }
        // ---- A tile: 128 rows x 32 k, f32 -> bf16 (16 elems/thread)
        {
            int row = tid >> 1;
            int kk  = (tid & 1) * 16;
            const float* src = A + (size_t)(mBlk + row) * E_ + k0 + kk;
            if (k0 + 32 < E_) __builtin_prefetch(src + 32, 0, 1);
            Pack16 tmp;
            #pragma unroll
            for (int q = 0; q < 4; ++q) {
                float4 f = *(const float4*)(src + q * 4);
                tmp.h[q*4+0] = (bf16)f.x; tmp.h[q*4+1] = (bf16)f.y;
                tmp.h[q*4+2] = (bf16)f.z; tmp.h[q*4+3] = (bf16)f.w;
            }
            bf16* dst = &lA[row * LDH + kk];
            *(uint4*)(dst)     = tmp.q[0];
            *(uint4*)(dst + 8) = tmp.q[1];
        }
        wait_async0();
        __syncthreads();

        // ---- fragment loads per ISA 16-bit layouts, then 4 WMMAs
        Frag fa[2], fb[2];
        #pragma unroll
        for (int ti = 0; ti < 2; ++ti) {
            const bf16* p = &lA[(mW * 32 + ti * 16 + lan) * LDH + laneHi * 8];
            fa[ti].q[0] = *(const uint4*)(p);
            fa[ti].q[1] = *(const uint4*)(p + 16);
        }
        #pragma unroll
        for (int to = 0; to < 2; ++to) {
            const bf16* p = &lW[(oW * 32 + to * 16 + lan) * LDH + laneHi * 8];
            fb[to].q[0] = *(const uint4*)(p);
            fb[to].q[1] = *(const uint4*)(p + 16);
        }
        #pragma unroll
        for (int ti = 0; ti < 2; ++ti)
            #pragma unroll
            for (int to = 0; to < 2; ++to)
                acc[ti][to] = __builtin_amdgcn_wmma_f32_16x16x32_bf16(
                    false, fa[ti].v, false, fb[to].v,
                    (short)0, acc[ti][to], false, false);
        __syncthreads();
    }

    // ---- epilogue: +bias, store Wh as bf16 (GEMM2 operand)
    #pragma unroll
    for (int to = 0; to < 2; ++to) {
        int o = oBlk + oW * 32 + to * 16 + lan;
        float bias = Wb[o];
        #pragma unroll
        for (int ti = 0; ti < 2; ++ti) {
            #pragma unroll
            for (int r = 0; r < 8; ++r) {
                int row = mBlk + mW * 32 + ti * 16 + r + laneHi * 8;
                Whbf[(size_t)row * OUT_ + o] = (bf16)(acc[ti][to][r] + bias);
            }
        }
    }
}

// ---------------------------------------------------------------------------
// Kernel 2: s1[r] = Wh[r,:].a1 ; s2[r] = Wh[r,:].a2  (one wave per row)
// ---------------------------------------------------------------------------
__global__ __launch_bounds__(256)
void gat_srow(const bf16* __restrict__ Whbf, const float* __restrict__ aw,
              float* __restrict__ s1, float* __restrict__ s2) {
    int row  = blockIdx.x * 8 + (threadIdx.x >> 5);
    int lane = threadIdx.x & 31;
    Pack16 w;
    w.q[0] = *(const uint4*)(Whbf + (size_t)row * OUT_ + lane * 8);
    float d1 = 0.f, d2 = 0.f;
    #pragma unroll
    for (int q = 0; q < 8; ++q) {
        float v = (float)w.h[q];
        int o = lane * 8 + q;
        d1 = fmaf(v, aw[o], d1);
        d2 = fmaf(v, aw[OUT_ + o], d2);
    }
    #pragma unroll
    for (int s = 16; s > 0; s >>= 1) {
        d1 += __shfl_xor(d1, s, 32);
        d2 += __shfl_xor(d2, s, 32);
    }
    if (lane == 0) { s1[row] = d1; s2[row] = d2; }
}

// ---------------------------------------------------------------------------
// Kernel 3: softmax stats. max_j lrelu(s1_i+s2_j) = lrelu(s1_i + max_j s2_j)
// (lrelu monotone), so only the exp-sum needs the j loop. One block per b.
// ---------------------------------------------------------------------------
__global__ __launch_bounds__(256)
void gat_stats(const float* __restrict__ s1, const float* __restrict__ s2,
               float* __restrict__ mrow, float* __restrict__ invd) {
    __shared__ float s2sh[N_];
    __shared__ float red[256];
    int b = blockIdx.x, tid = threadIdx.x;
    float mx = -1e30f;
    for (int j = tid; j < N_; j += 256) {
        float v = s2[b * N_ + j];
        s2sh[j] = v;
        mx = fmaxf(mx, v);
    }
    red[tid] = mx;
    __syncthreads();
    for (int s = 128; s > 0; s >>= 1) {
        if (tid < s) red[tid] = fmaxf(red[tid], red[tid + s]);
        __syncthreads();
    }
    float s2max = red[0];
    #pragma unroll
    for (int t = 0; t < 4; ++t) {
        int i = tid + t * 256;
        float a = s1[b * N_ + i];
        float m = lrelu(a + s2max);
        float sum = 0.f;
        for (int j = 0; j < N_; ++j)
            sum += __expf(lrelu(a + s2sh[j]) - m);
        mrow[b * N_ + i] = m;
        invd[b * N_ + i] = 1.0f / sum;
    }
}

// ---------------------------------------------------------------------------
// Kernel 4: pooled[b,o] += sum_i elu( sum_j P[i,j]*Wh[b,j,o] )
// P regenerated on the fly as bf16 tiles (1/denom folded in, linear in GEMM).
// WG tile 128(i) x 64(o), loop j in steps of 32; h' never hits memory.
// s2 row block preloaded via async global->LDS.
// ---------------------------------------------------------------------------
__global__ __launch_bounds__(256)
void gat_gemm2(const bf16* __restrict__ Whbf, const float* __restrict__ s1,
               const float* __restrict__ s2, const float* __restrict__ mrow,
               const float* __restrict__ invd, float* __restrict__ pooled) {
    __shared__ bf16  lP[128 * LDH];
    __shared__ bf16  lB[64 * LDH];
    __shared__ float s2sh[N_];
    __shared__ float s1sh[128], msh[128], ish[128];

    const int b      = blockIdx.z;
    const int iBlk   = blockIdx.x * 128;
    const int oBlk   = blockIdx.y * 64;
    const int tid    = threadIdx.x;
    const int lane   = tid & 31;
    const int wave   = tid >> 5;
    const int mW     = wave & 3;
    const int oW     = wave >> 2;
    const int lan    = lane & 15;
    const int laneHi = lane >> 4;

    // s2[b, 0:1024] -> LDS, 4 floats (16B) per lane, async path
    async_copy_b128(s2 + b * N_ + tid * 4, &s2sh[tid * 4]);
    if (tid < 128) {
        s1sh[tid] = s1[b * N_ + iBlk + tid];
        msh[tid]  = mrow[b * N_ + iBlk + tid];
        ish[tid]  = invd[b * N_ + iBlk + tid];
    }
    wait_async0();
    __syncthreads();

    v8f acc[2][2] = {};

    for (int j0 = 0; j0 < N_; j0 += 32) {
        // ---- build P tile (128 i x 32 j) in bf16: 16 probs/thread
        {
            int i  = tid >> 1;
            int jj = (tid & 1) * 16;
            float a = s1sh[i], m = msh[i], iv = ish[i];
            Pack16 tmp;
            #pragma unroll
            for (int q = 0; q < 16; ++q) {
                float x = a + s2sh[j0 + jj + q];
                tmp.h[q] = (bf16)(__expf(lrelu(x) - m) * iv);
            }
            bf16* dst = &lP[i * LDH + jj];
            *(uint4*)(dst)     = tmp.q[0];
            *(uint4*)(dst + 8) = tmp.q[1];
        }
        // ---- stage Wh tile transposed: lB[o][j] (coalesced global read)
        {
            int jj = tid >> 3;
            int op = (tid & 7) * 8;
            Pack16 w;
            w.q[0] = *(const uint4*)(Whbf +
                        ((size_t)(b * N_ + j0 + jj)) * OUT_ + oBlk + op);
            #pragma unroll
            for (int q = 0; q < 8; ++q) lB[(op + q) * LDH + jj] = w.h[q];
        }
        __syncthreads();

        Frag fa[2], fb[2];
        #pragma unroll
        for (int ti = 0; ti < 2; ++ti) {
            const bf16* p = &lP[(mW * 32 + ti * 16 + lan) * LDH + laneHi * 8];
            fa[ti].q[0] = *(const uint4*)(p);
            fa[ti].q[1] = *(const uint4*)(p + 16);
        }
        #pragma unroll
        for (int to = 0; to < 2; ++to) {
            const bf16* p = &lB[(oW * 32 + to * 16 + lan) * LDH + laneHi * 8];
            fb[to].q[0] = *(const uint4*)(p);
            fb[to].q[1] = *(const uint4*)(p + 16);
        }
        #pragma unroll
        for (int ti = 0; ti < 2; ++ti)
            #pragma unroll
            for (int to = 0; to < 2; ++to)
                acc[ti][to] = __builtin_amdgcn_wmma_f32_16x16x32_bf16(
                    false, fa[ti].v, false, fb[to].v,
                    (short)0, acc[ti][to], false, false);
        __syncthreads();
    }

    // ---- fused elu + mean-pool reduction over this wave's 32 rows
    #pragma unroll
    for (int to = 0; to < 2; ++to) {
        float s = 0.f;
        #pragma unroll
        for (int ti = 0; ti < 2; ++ti)
            #pragma unroll
            for (int r = 0; r < 8; ++r) {
                float x = acc[ti][to][r];
                s += (x > 0.f) ? x : (__expf(x) - 1.0f);
            }
        s += __shfl_xor(s, 16, 32);   // combine the two lane-groups (rows)
        if (lane < 16)
            atomicAdd(&pooled[b * OUT_ + oBlk + oW * 32 + to * 16 + lan], s);
    }
}

// ---------------------------------------------------------------------------
// Kernel 5: out[b,k] = (pooled[b,:]/N) . mlp_w[k,:] + mlp_b[k]   (64 outputs)
// ---------------------------------------------------------------------------
__global__ __launch_bounds__(64)
void gat_head(const float* __restrict__ pooled, const float* __restrict__ mw,
              const float* __restrict__ mb, float* __restrict__ out) {
    int t = threadIdx.x;
    if (t >= 64) return;
    int b = t >> 1, k = t & 1;
    float s = 0.f;
    for (int o = 0; o < OUT_; ++o)
        s = fmaf(pooled[b * OUT_ + o], mw[k * OUT_ + o], s);
    out[b * 2 + k] = s * (1.0f / N_) + mb[k];
}

// ---------------------------------------------------------------------------
extern "C" void kernel_launch(void* const* d_in, const int* in_sizes, int n_in,
                              void* d_out, int out_size, void* d_ws, size_t ws_size,
                              hipStream_t stream) {
    (void)in_sizes; (void)n_in; (void)out_size; (void)ws_size;
    const float* sent = (const float*)d_in[0];
    // d_in[1] = batch_e : dead code (event_mask in {0.9, 0.1}, both > 0)
    const float* Ww = (const float*)d_in[2];
    const float* Wb = (const float*)d_in[3];
    const float* aw = (const float*)d_in[4];
    const float* mw = (const float*)d_in[5];
    const float* mb = (const float*)d_in[6];
    float* out = (float*)d_out;

    char* ws = (char*)d_ws;                                // ~17.1 MB total
    bf16*  Whbf   = (bf16*) (ws);                          // 16 MB
    bf16*  Wbf    = (bf16*) (ws + 16777216);               // 384 KB
    float* s1     = (float*)(ws + 17170432);               // 128 KB
    float* s2     = (float*)(ws + 17301504);               // 128 KB
    float* mrow   = (float*)(ws + 17432576);               // 128 KB
    float* invd   = (float*)(ws + 17563648);               // 128 KB
    float* pooled = (float*)(ws + 17694720);               // 32 KB

    gat_prep <<<768, 256, 0, stream>>>(Ww, Wbf, pooled);
    gat_gemm1<<<dim3(256, 4), 256, 0, stream>>>(sent, Wbf, Wb, Whbf);
    gat_srow <<<4096, 256, 0, stream>>>(Whbf, aw, s1, s2);
    gat_stats<<<32, 256, 0, stream>>>(s1, s2, mrow, invd);
    gat_gemm2<<<dim3(8, 4, 32), 256, 0, stream>>>(Whbf, s1, s2, mrow, invd, pooled);
    gat_head <<<1, 64, 0, stream>>>(pooled, mw, mb, out);
}
